// PinSAGEModel_32564442038561
// MI455X (gfx1250) — compile-verified, matrix operand order checked by reading, not run on hardware
//
#include <hip/hip_runtime.h>
#include <hip/hip_bf16.h>

#define N0 331776
#define NN1 36864
#define NN2 4096
#define E0 294912
#define E1 32768
#define EPE 4096
#define FR 128
#define DD 256
#define HH 512

typedef __attribute__((ext_vector_type(16))) __bf16 v16bf;
typedef __attribute__((ext_vector_type(8)))  float  v8f;

struct U8x4 { uint4 a, b; };   // 32 bytes == v16bf

// LDS tile geometry: padded k-stride 72 elems (144 B) -> ds_load_b128 bank-conflict free
#define KSTRIDE 72
#define A_ELEMS (64 * KSTRIDE)          // 4608
#define B_ELEMS (128 * KSTRIDE)         // 9216
#define STAGE_ELEMS (A_ELEMS + B_ELEMS) // 13824

__device__ __forceinline__ void async_b128(const __bf16* g, unsigned ldsoff) {
    // ASYNCcnt-tracked DMA of 16B/lane from global to LDS (gfx1250).
    asm volatile("global_load_async_to_lds_b128 %0, %1, off"
                 :: "v"(ldsoff), "v"(g) : "memory");
}

template<int N>
__device__ __forceinline__ void wait_async() {
    asm volatile("s_wait_asynccnt %0" :: "n"(N) : "memory");
}

// Read one 16x32 bf16 fragment from an LDS tile (rows x KSTRIDE, row-major).
// ISA 16-bit 16x32 layout: lane L -> row = L&15, g = L>>4;
// elems 0..7 = K = g*8+0..7 ; elems 8..15 = K = 16+g*8+0..7 (two 16B ds loads)
__device__ __forceinline__ v16bf lds_frag(const __bf16* buf, int r0, int kb, int lane) {
    const __bf16* p = buf + (r0 + (lane & 15)) * KSTRIDE + kb + ((lane >> 4) << 3);
    U8x4 t;
    t.a = *(const uint4*)(p);
    t.b = *(const uint4*)(p + 16);
    return __builtin_bit_cast(v16bf, t);
}

__device__ __forceinline__ v8f wmma_bf16(v16bf a, v16bf b, v8f c) {
    return __builtin_amdgcn_wmma_f32_16x16x32_bf16(false, a, false, b, (short)0, c, false, false);
}

// Issue one stage (K-slab of 64) of async copies: A 64x64 (2 vec/thread), B 128x64 (4 vec/thread).
__device__ __forceinline__ void stage_issue(const __bf16* __restrict__ A,
                                            const __bf16* __restrict__ Bt, int K,
                                            int gr0, int gc0, int kb,
                                            __bf16* sA, __bf16* sB, int tid) {
#pragma unroll
    for (int i = 0; i < 2; ++i) {
        int v = tid + i * 256; int row = v >> 3; int kc = (v & 7) << 3;
        async_b128(A + (size_t)(gr0 + row) * K + kb + kc,
                   (unsigned)(size_t)(sA + row * KSTRIDE + kc));
    }
#pragma unroll
    for (int i = 0; i < 4; ++i) {
        int v = tid + i * 256; int row = v >> 3; int kc = (v & 7) << 3;
        async_b128(Bt + (size_t)(gc0 + row) * K + kb + kc,
                   (unsigned)(size_t)(sB + row * KSTRIDE + kc));
    }
}

// C[M,N] = A[M,K](bf16) * Bt[N,K]^T(bf16) + bias, optional relu; fp32 and/or bf16 out.
// 256 threads = 8 waves (2x4); wave tile 32x32; macro tile 64x128; K staged 64 at a
// time through double-buffered LDS filled with global_load_async_to_lds_b128.
// Requires M%64==0, N%128==0, K%64==0 (all call sites satisfy this).
template<bool RELU>
__global__ __launch_bounds__(256)
void gemm_bf16_wmma(const __bf16* __restrict__ A, const __bf16* __restrict__ Bt,
                    const float* __restrict__ bias, float* __restrict__ Cf,
                    __bf16* __restrict__ Cb, int M, int N, int K)
{
    __shared__ __bf16 smem[2 * STAGE_ELEMS];   // 55296 B
    const int tid  = threadIdx.x;
    const int lane = tid & 31;
    const int wid  = tid >> 5;
    const int r0l  = (wid >> 2) * 32;          // wave row within macro tile
    const int c0l  = (wid & 3)  * 32;          // wave col within macro tile
    const int gr0  = blockIdx.y * 64;
    const int gc0  = blockIdx.x * 128;
    const int row0 = gr0 + r0l;
    const int col0 = gc0 + c0l;

    v8f acc[2][2];
#pragma unroll
    for (int mi = 0; mi < 2; ++mi)
#pragma unroll
        for (int nj = 0; nj < 2; ++nj) {
            float bv = bias ? bias[col0 + nj * 16 + (lane & 15)] : 0.0f;
#pragma unroll
            for (int r = 0; r < 8; ++r) acc[mi][nj][r] = bv;
        }

    const int NS = K >> 6;
    stage_issue(A, Bt, K, gr0, gc0, 0, smem, smem + A_ELEMS, tid);

    for (int ks = 0; ks < NS; ++ks) {
        if (ks + 1 < NS) {   // prefetch next slab into the other buffer
            __bf16* nb = smem + ((ks + 1) & 1) * STAGE_ELEMS;
            stage_issue(A, Bt, K, gr0, gc0, (ks + 1) << 6, nb, nb + A_ELEMS, tid);
            wait_async<6>(); // 6 newer in flight -> current stage's 6 have landed
        } else {
            wait_async<0>();
        }
        __syncthreads();     // all waves' current stage visible

        const __bf16* sA = smem + (ks & 1) * STAGE_ELEMS;
        const __bf16* sB = sA + A_ELEMS;
#pragma unroll
        for (int kk = 0; kk < 64; kk += 32) {
            v16bf a0 = lds_frag(sA, r0l,      kk, lane);
            v16bf a1 = lds_frag(sA, r0l + 16, kk, lane);
            v16bf b0 = lds_frag(sB, c0l,      kk, lane);
            v16bf b1 = lds_frag(sB, c0l + 16, kk, lane);
            acc[0][0] = wmma_bf16(a0, b0, acc[0][0]);
            acc[0][1] = wmma_bf16(a0, b1, acc[0][1]);
            acc[1][0] = wmma_bf16(a1, b0, acc[1][0]);
            acc[1][1] = wmma_bf16(a1, b1, acc[1][1]);
        }
        __syncthreads();     // everyone done reading this buffer before next overwrite
    }

#pragma unroll
    for (int mi = 0; mi < 2; ++mi)
#pragma unroll
        for (int nj = 0; nj < 2; ++nj) {
            const int col = col0 + nj * 16 + (lane & 15);
#pragma unroll
            for (int r = 0; r < 8; ++r) {
                const int row = row0 + mi * 16 + ((lane >> 4) << 3) + r;
                float v = acc[mi][nj][r];
                if (RELU) v = fmaxf(v, 0.0f);
                size_t idx = (size_t)row * N + col;
                if (Cf) Cf[idx] = v;
                if (Cb) Cb[idx] = (__bf16)v;
            }
        }
}

__global__ void fill_zero(float* __restrict__ p, size_t n) {
    size_t t = (size_t)blockIdx.x * blockDim.x + threadIdx.x;
    if (t < n) p[t] = 0.0f;
}

__global__ void cvt_bf16(const float* __restrict__ in, __bf16* __restrict__ out, size_t n) {
    size_t t = (size_t)blockIdx.x * blockDim.x + threadIdx.x;
    if (t < n) out[t] = (__bf16)in[t];
}

// W[K,N] fp32 -> Wt[N,K] bf16
__global__ void tcvt_bf16(const float* __restrict__ W, __bf16* __restrict__ Wt, int K, int N) {
    int t = blockIdx.x * blockDim.x + threadIdx.x;
    if (t >= K * N) return;
    int k = t / N, n = t % N;
    Wt[(size_t)n * K + k] = (__bf16)W[t];
}

__global__ void edge_wsum(const float* __restrict__ w, const int* __restrict__ dst,
                          float* __restrict__ wsum, int E) {
    int e = blockIdx.x * blockDim.x + threadIdx.x;
    if (e < E) atomicAdd(&wsum[dst[e]], w[e]);
}

// m[dst] += w[e] * n[src]   (H = 512, 4 elems per thread)
__global__ void edge_msgs(const __bf16* __restrict__ nsrc, const int* __restrict__ src,
                          const int* __restrict__ dst, const float* __restrict__ w,
                          float* __restrict__ m, int E) {
    size_t t = (size_t)blockIdx.x * blockDim.x + threadIdx.x;
    if (t >= (size_t)E * 128) return;
    int e = (int)(t >> 7);
    int c = (int)(t & 127) << 2;
    float we = w[e];
    const __bf16* np = nsrc + (size_t)src[e] * HH + c;
    float*        mp = m    + (size_t)dst[e] * HH + c;
#pragma unroll
    for (int i = 0; i < 4; ++i)
        atomicAdd(mp + i, (float)np[i] * we);
}

// xcat[r, 0:512] = m[r]/max(ws[r],1) ; xcat[r, 512:768] = hdst[r]  (bf16 out)
__global__ void build_cat(const float* __restrict__ m, const float* __restrict__ wsum,
                          const __bf16* __restrict__ hdst, __bf16* __restrict__ xcat, int R) {
    size_t t = (size_t)blockIdx.x * blockDim.x + threadIdx.x;
    if (t >= (size_t)R * (HH + DD)) return;
    int r = (int)(t / (HH + DD));
    int c = (int)(t % (HH + DD));
    float o;
    if (c < HH) o = m[(size_t)r * HH + c] / fmaxf(wsum[r], 1.0f);
    else        o = (float)hdst[(size_t)r * DD + (c - HH)];
    xcat[t] = (__bf16)o;
}

// row-wise l2 normalize (256 cols), optional skip-add, fp32/bf16 outs. 1 wave per row.
__global__ __launch_bounds__(256)
void l2norm_kernel(const float* __restrict__ in, __bf16* __restrict__ outb,
                   float* __restrict__ outf, const float* __restrict__ skip, int R) {
    const int lane = threadIdx.x & 31;
    const int row  = blockIdx.x * 8 + (threadIdx.x >> 5);
    if (row >= R) return;
    const float* p = in + (size_t)row * DD;
    float v[8]; float s = 0.0f;
#pragma unroll
    for (int i = 0; i < 8; ++i) { v[i] = p[i * 32 + lane]; s += v[i] * v[i]; }
#pragma unroll
    for (int off = 16; off > 0; off >>= 1) s += __shfl_xor(s, off, 32);
    const float inv = (s > 0.0f) ? (1.0f / sqrtf(s)) : 1.0f;
#pragma unroll
    for (int i = 0; i < 8; ++i) {
        const int c = i * 32 + lane;
        float o = v[i] * inv;
        if (skip) o += skip[(size_t)row * DD + c];
        if (outf) outf[(size_t)row * DD + c] = o;
        if (outb) outb[(size_t)row * DD + c] = (__bf16)o;
    }
}

__global__ void score_kernel(const float* __restrict__ h, const int* __restrict__ ps,
                             const int* __restrict__ pd, const int* __restrict__ ns,
                             const int* __restrict__ nd, const int* __restrict__ nid,
                             const float* __restrict__ bias, float* __restrict__ out) {
    int t = blockIdx.x * blockDim.x + threadIdx.x;
    if (t >= 2 * EPE) return;
    bool neg = t >= EPE;
    int e = neg ? t - EPE : t;
    int u = (neg ? ns : ps)[e];
    int v = (neg ? nd : pd)[e];
    const float4* hu = (const float4*)(h + (size_t)u * DD);
    const float4* hv = (const float4*)(h + (size_t)v * DD);
    float acc = 0.0f;
#pragma unroll 4
    for (int i = 0; i < DD / 4; ++i) {
        float4 a = hu[i], b = hv[i];
        acc += a.x * b.x + a.y * b.y + a.z * b.z + a.w * b.w;
    }
    out[t] = acc + bias[nid[u]] + bias[nid[v]];
}

static inline int cdiv(long long a, int b) { return (int)((a + b - 1) / b); }

extern "C" void kernel_launch(void* const* d_in, const int* in_sizes, int n_in,
                              void* d_out, int out_size, void* d_ws, size_t ws_size,
                              hipStream_t stream) {
    (void)in_sizes; (void)n_in; (void)out_size; (void)ws_size;
    const float* feat0    = (const float*)d_in[0];
    const float* feat_dst = (const float*)d_in[1];
    const float* w0       = (const float*)d_in[2];
    const float* w1       = (const float*)d_in[3];
    const float* W_proj   = (const float*)d_in[4];
    const float* b_proj   = (const float*)d_in[5];
    const float* Q1       = (const float*)d_in[6];
    const float* bq1      = (const float*)d_in[7];
    const float* W1       = (const float*)d_in[8];
    const float* bw1      = (const float*)d_in[9];
    const float* Q2       = (const float*)d_in[10];
    const float* bq2      = (const float*)d_in[11];
    const float* W2       = (const float*)d_in[12];
    const float* bw2      = (const float*)d_in[13];
    const float* bias     = (const float*)d_in[14];
    const int* b0s = (const int*)d_in[15];
    const int* b0d = (const int*)d_in[16];
    const int* b1s = (const int*)d_in[17];
    const int* b1d = (const int*)d_in[18];
    const int* pos_s = (const int*)d_in[19];
    const int* pos_d = (const int*)d_in[20];
    const int* neg_s = (const int*)d_in[21];
    const int* neg_d = (const int*)d_in[22];
    const int* nid   = (const int*)d_in[23];
    float* out = (float*)d_out;

    // ---- workspace carve ----
    char* base = (char*)d_ws;
    size_t off = 0;
    auto take = [&](size_t bytes) -> void* {
        void* p = base + off;
        off += bytes;
        off = (off + 255) & ~(size_t)255;
        return p;
    };
    __bf16* feat0b  = (__bf16*)take((size_t)N0 * FR * 2);
    __bf16* fdstb   = (__bf16*)take((size_t)NN2 * FR * 2);
    __bf16* WprojT  = (__bf16*)take((size_t)DD * FR * 2);
    __bf16* Q1T     = (__bf16*)take((size_t)HH * DD * 2);
    __bf16* W1T     = (__bf16*)take((size_t)DD * (DD + HH) * 2);
    __bf16* Q2T     = (__bf16*)take((size_t)HH * DD * 2);
    __bf16* W2T     = (__bf16*)take((size_t)DD * (DD + HH) * 2);
    __bf16* h0b     = (__bf16*)take((size_t)N0 * DD * 2);
    __bf16* n1b     = (__bf16*)take((size_t)N0 * HH * 2);
    float*  m1      = (float*)take((size_t)NN1 * HH * 4);
    float*  ws1     = (float*)take((size_t)NN1 * 4);
    __bf16* xcat1   = (__bf16*)take((size_t)NN1 * (HH + DD) * 2);
    float*  z1raw   = (float*)take((size_t)NN1 * DD * 4);
    __bf16* z1b     = (__bf16*)take((size_t)NN1 * DD * 2);
    __bf16* n2b     = (__bf16*)take((size_t)NN1 * HH * 2);
    float*  m2      = (float*)take((size_t)NN2 * HH * 4);
    float*  ws2     = (float*)take((size_t)NN2 * 4);
    __bf16* xcat2   = (__bf16*)take((size_t)NN2 * (HH + DD) * 2);
    float*  z2raw   = (float*)take((size_t)NN2 * DD * 4);
    float*  projdst = (float*)take((size_t)NN2 * DD * 4);
    float*  hitem   = (float*)take((size_t)NN2 * DD * 4);

    const int T = 256;
    // ---- conversions ----
    cvt_bf16<<<cdiv((long long)N0 * FR, T), T, 0, stream>>>(feat0, feat0b, (size_t)N0 * FR);
    cvt_bf16<<<cdiv((long long)NN2 * FR, T), T, 0, stream>>>(feat_dst, fdstb, (size_t)NN2 * FR);
    tcvt_bf16<<<cdiv(FR * DD, T), T, 0, stream>>>(W_proj, WprojT, FR, DD);
    tcvt_bf16<<<cdiv(DD * HH, T), T, 0, stream>>>(Q1, Q1T, DD, HH);
    tcvt_bf16<<<cdiv((DD + HH) * DD, T), T, 0, stream>>>(W1, W1T, DD + HH, DD);
    tcvt_bf16<<<cdiv(DD * HH, T), T, 0, stream>>>(Q2, Q2T, DD, HH);
    tcvt_bf16<<<cdiv((DD + HH) * DD, T), T, 0, stream>>>(W2, W2T, DD + HH, DD);
    // ---- zero accumulators (must happen every call: graph replays) ----
    fill_zero<<<cdiv((long long)NN1 * HH, T), T, 0, stream>>>(m1, (size_t)NN1 * HH);
    fill_zero<<<cdiv(NN1, T), T, 0, stream>>>(ws1, (size_t)NN1);
    fill_zero<<<cdiv((long long)NN2 * HH, T), T, 0, stream>>>(m2, (size_t)NN2 * HH);
    fill_zero<<<cdiv(NN2, T), T, 0, stream>>>(ws2, (size_t)NN2);

    // ---- h0 = feat0 @ W_proj + b_proj (bf16 out) ----
    gemm_bf16_wmma<false><<<dim3(DD / 128, N0 / 64), T, 0, stream>>>(
        feat0b, WprojT, b_proj, nullptr, h0b, N0, DD, FR);
    // ---- n1 = relu(h0 @ Q1 + bq1) (bf16 out) ----
    gemm_bf16_wmma<true><<<dim3(HH / 128, N0 / 64), T, 0, stream>>>(
        h0b, Q1T, bq1, nullptr, n1b, N0, HH, DD);
    // ---- layer1 segment sums ----
    edge_wsum<<<cdiv(E0, T), T, 0, stream>>>(w0, b0d, ws1, E0);
    edge_msgs<<<cdiv((long long)E0 * 128, T), T, 0, stream>>>(n1b, b0s, b0d, w0, m1, E0);
    build_cat<<<cdiv((long long)NN1 * (HH + DD), T), T, 0, stream>>>(m1, ws1, h0b, xcat1, NN1);
    // ---- z1 = l2norm(relu(cat @ W1 + bw1)) ----
    gemm_bf16_wmma<true><<<dim3(DD / 128, NN1 / 64), T, 0, stream>>>(
        xcat1, W1T, bw1, z1raw, nullptr, NN1, DD, DD + HH);
    l2norm_kernel<<<cdiv(NN1, 8), T, 0, stream>>>(z1raw, z1b, nullptr, nullptr, NN1);
    // ---- n2 = relu(z1 @ Q2 + bq2) ----
    gemm_bf16_wmma<true><<<dim3(HH / 128, NN1 / 64), T, 0, stream>>>(
        z1b, Q2T, bq2, nullptr, n2b, NN1, HH, DD);
    // ---- layer2 segment sums ----
    edge_wsum<<<cdiv(E1, T), T, 0, stream>>>(w1, b1d, ws2, E1);
    edge_msgs<<<cdiv((long long)E1 * 128, T), T, 0, stream>>>(n2b, b1s, b1d, w1, m2, E1);
    build_cat<<<cdiv((long long)NN2 * (HH + DD), T), T, 0, stream>>>(m2, ws2, z1b, xcat2, NN2);
    // ---- z2raw = relu(cat @ W2 + bw2) ----
    gemm_bf16_wmma<true><<<dim3(DD / 128, NN2 / 64), T, 0, stream>>>(
        xcat2, W2T, bw2, z2raw, nullptr, NN2, DD, DD + HH);
    // ---- projdst = feat_dst @ W_proj + b_proj ----
    gemm_bf16_wmma<false><<<dim3(DD / 128, NN2 / 64), T, 0, stream>>>(
        fdstb, WprojT, b_proj, projdst, nullptr, NN2, DD, FR);
    // ---- h_item = l2norm(z2raw) + projdst ----
    l2norm_kernel<<<cdiv(NN2, 8), T, 0, stream>>>(z2raw, nullptr, hitem, projdst, NN2);
    // ---- scores ----
    score_kernel<<<cdiv(2 * EPE, T), T, 0, stream>>>(hitem, pos_s, pos_d, neg_s, neg_d,
                                                     nid, bias, out);
}
